// ShapelyMixer_83124797047295
// MI455X (gfx1250) — matrix operand order repaired and checked by reading, not compiled, not run on hardware
//
#include <hip/hip_runtime.h>
#include <stdint.h>

// Problem constants (fixed by reference): T=32,B=32 -> BT=1024, S=128, N=8, A=10, E=64, Ss=32
#define BT    1024
#define SDIM  128
#define NAG   8
#define AACT  10
#define EH    64
#define SSAMP 32
#define K1    80      // N*A
#define K1PAD 96      // padded to 3 x 32

typedef __attribute__((ext_vector_type(16))) _Float16 v16h;
typedef __attribute__((ext_vector_type(8)))  float    v8f;
typedef __attribute__((ext_vector_type(8)))  uint32_t v8u;

// ---------------------------------------------------------------------------
// Threefry2x32 (JAX-compatible) for reproducing jax.random.uniform
// ---------------------------------------------------------------------------
static __device__ __forceinline__ uint32_t rotl32(uint32_t x, uint32_t d) {
  return (x << d) | (x >> (32u - d));
}

static __device__ __forceinline__ void threefry2x32(uint32_t k0, uint32_t k1,
                                                    uint32_t c0, uint32_t c1,
                                                    uint32_t& o0, uint32_t& o1) {
  uint32_t ks0 = k0, ks1 = k1, ks2 = 0x1BD11BDAu ^ k0 ^ k1;
  uint32_t x0 = c0 + ks0, x1 = c1 + ks1;
#define TF_R(r) { x0 += x1; x1 = rotl32(x1, r); x1 ^= x0; }
  TF_R(13) TF_R(15) TF_R(26) TF_R(6)   x0 += ks1; x1 += ks2 + 1u;
  TF_R(17) TF_R(29) TF_R(16) TF_R(24)  x0 += ks2; x1 += ks0 + 2u;
  TF_R(13) TF_R(15) TF_R(26) TF_R(6)   x0 += ks0; x1 += ks1 + 3u;
  TF_R(17) TF_R(29) TF_R(16) TF_R(24)  x0 += ks1; x1 += ks2 + 4u;
  TF_R(13) TF_R(15) TF_R(26) TF_R(6)   x0 += ks2; x1 += ks0 + 5u;
#undef TF_R
  o0 = x0; o1 = x1;
}

static __device__ __forceinline__ float jax_uniform_elem(uint32_t k1, uint32_t e) {
  const uint32_t HALF = (BT * SSAMP * NAG) / 2u; // 131072 (even total)
  uint32_t o0, o1, bits;
  if (e < HALF) { threefry2x32(0u, k1, e, e + HALF, o0, o1); bits = o0; }
  else          { threefry2x32(0u, k1, e - HALF, e, o0, o1); bits = o1; }
  return __uint_as_float((bits >> 9) | 0x3F800000u) - 1.0f;
}

// ---------------------------------------------------------------------------
// Kernel 1: permutations. One thread per (bt, s): 8 uniforms, stable argsort,
// store p2a (position->agent) and a2p (agent->position) as bytes.
// ---------------------------------------------------------------------------
__global__ __launch_bounds__(256) void perm_kernel(const int* rng,
                                                   uint8_t* p2a_g, uint8_t* a2p_g) {
  int idx = blockIdx.x * blockDim.x + threadIdx.x;   // 0 .. BT*SSAMP-1
  if (idx >= BT * SSAMP) return;
  uint32_t k1 = (uint32_t)rng[0];
  float r[NAG];
#pragma unroll
  for (int n = 0; n < NAG; ++n) r[n] = jax_uniform_elem(k1, (uint32_t)(idx * NAG + n));
  int ord[NAG];
#pragma unroll
  for (int n = 0; n < NAG; ++n) ord[n] = n;
  // stable insertion sort ascending (strict > keeps ties in original order)
  for (int i = 1; i < NAG; ++i) {
    int oi = ord[i]; float vq = r[oi]; int j = i - 1;
    while (j >= 0 && r[ord[j]] > vq) { ord[j + 1] = ord[j]; --j; }
    ord[j + 1] = oi;
  }
  uint8_t inv[NAG];
#pragma unroll
  for (int k = 0; k < NAG; ++k) inv[ord[k]] = (uint8_t)k;
#pragma unroll
  for (int n = 0; n < NAG; ++n) {
    p2a_g[idx * NAG + n] = (uint8_t)ord[n];
    a2p_g[idx * NAG + n] = inv[n];
  }
}

// ---------------------------------------------------------------------------
// Kernel 2: per-batch-row state projections.
// spb1[b][e] = states[b] @ W1[:128] + b1     (used as WMMA C-init)
// v[b]       = relu(states[b] @ Wv1 + bv1) @ Wv2 + bv2
// ---------------------------------------------------------------------------
__global__ __launch_bounds__(64) void state_kernel(const float* states, const float* W1,
                                                   const float* b1, const float* Wv1,
                                                   const float* bv1, const float* Wv2,
                                                   const float* bv2, float* spb1_g,
                                                   float* v_g) {
  int b = blockIdx.x;          // 0..BT-1
  int e = threadIdx.x;         // 0..63
  __shared__ float st[SDIM];
  __shared__ float red[EH];
  for (int i = e; i < SDIM; i += EH) st[i] = states[b * SDIM + i];
  __syncthreads();
  float s1 = b1[e], s2 = bv1[e];
  for (int k = 0; k < SDIM; ++k) {
    float sv = st[k];
    s1 += sv * W1[k * EH + e];
    s2 += sv * Wv1[k * EH + e];
  }
  spb1_g[b * EH + e] = s1;
  float hv = s2 > 0.f ? s2 : 0.f;
  red[e] = hv * Wv2[e];
  __syncthreads();
  if (e == 0) {
    float acc = bv2[0];
    for (int k = 0; k < EH; ++k) acc += red[k];
    v_g[b] = acc;
  }
}

// ---------------------------------------------------------------------------
// Kernel 3: pack W1[128:208] (pad K to 96) and W2 (K=64) into f16 B-fragments.
// B k order chosen here (contiguous, A uses the same; not the ISA table order): lane = half*16 + col, dword v packs
// K = half*16 + 2v, 2v+1 (plus kstep*32).
// Storage: [kstep][ntile][lane][8 dwords] so each lane loads 8 contiguous dwords.
// ---------------------------------------------------------------------------
__global__ __launch_bounds__(256) void pack_kernel(const float* W1, const float* W2,
                                                   uint32_t* w1pack, uint32_t* w2pack) {
  int id = blockIdx.x * blockDim.x + threadIdx.x;  // 0..639
  if (id >= 640) return;
  union { _Float16 hh[2]; uint32_t u; } pk;
  if (id < 384) {              // layer 1: 3 ksteps x 4 ntiles x 32 lanes
    int lane = id & 31, nt = (id >> 5) & 3, ks = id >> 7;
    int h = lane >> 4, c = nt * 16 + (lane & 15);
#pragma unroll
    for (int v = 0; v < 8; ++v) {
      int k0 = ks * 32 + h * 16 + 2 * v;
      float f0 = (k0     < K1) ? W1[(SDIM + k0    ) * EH + c] : 0.f;
      float f1 = (k0 + 1 < K1) ? W1[(SDIM + k0 + 1) * EH + c] : 0.f;
      pk.hh[0] = (_Float16)f0; pk.hh[1] = (_Float16)f1;
      w1pack[id * 8 + v] = pk.u;
    }
  } else {                     // layer 2: 2 ksteps x 4 ntiles x 32 lanes
    int t = id - 384;
    int lane = t & 31, nt = (t >> 5) & 3, ks = t >> 7;
    int h = lane >> 4, c = nt * 16 + (lane & 15);
#pragma unroll
    for (int v = 0; v < 8; ++v) {
      int k0 = ks * 32 + h * 16 + 2 * v;   // always < 64
      pk.hh[0] = (_Float16)W2[k0 * EH + c];
      pk.hh[1] = (_Float16)W2[(k0 + 1) * EH + c];
      w2pack[t * 8 + v] = pk.u;
    }
  }
}

// ---------------------------------------------------------------------------
// Kernel 4: main fused mixer. One block per batch row b (1024 blocks).
// 4 waves x 4 M-tiles; M-tile = 16 rows = 2 samples x 8 agents.
// Layer1: 12x v_wmma_f32_16x16x32_f16 (K=96), Layer2: 8x (K=64).
// ---------------------------------------------------------------------------
__global__ __launch_bounds__(128) void mixer_main(
    const float* agent_qs,       // (BT, A, N) flat: [b*80 + a*8 + n]
    const float* b2, const float* Wa, const float* ba,
    const uint8_t* p2a_g, const uint8_t* a2p_g,
    const float* spb1_g, const float* v_g,
    const v8u* w1pack, const v8u* w2pack,
    float* out) {
  __shared__ v8u w1f[12 * 32];          // 12 KB
  __shared__ v8u w2f[8 * 32];           //  8 KB
  __shared__ _Float16 h1s[4 * 16 * EH]; //  8 KB per-wave scratch (transpose bounce)
  __shared__ float aq[K1];              // agent_qs_flat[b, j, a] at aq[a*8 + j]
  __shared__ float spb1[EH];
  __shared__ float b2s[EH];
  __shared__ float was[EH];
  __shared__ uint8_t p2a_s[SSAMP * NAG];
  __shared__ uint8_t a2p_s[SSAMP * NAG];
  __shared__ float out_acc[NAG];

  const int b = blockIdx.x;
  const int tid = threadIdx.x;

  for (int i = tid; i < 12 * 32; i += 128) w1f[i] = w1pack[i];
  for (int i = tid; i < 8 * 32;  i += 128) w2f[i] = w2pack[i];
  if (tid < K1) aq[tid] = agent_qs[b * K1 + tid];
  if (tid < EH) { spb1[tid] = spb1_g[b * EH + tid]; b2s[tid] = b2[tid]; was[tid] = Wa[tid]; }
  for (int i = tid; i < SSAMP * NAG; i += 128) {
    p2a_s[i] = p2a_g[b * SSAMP * NAG + i];
    a2p_s[i] = a2p_g[b * SSAMP * NAG + i];
  }
  if (tid < NAG) out_acc[tid] = 0.f;
  __syncthreads();

  const int wave = tid >> 5;
  const int lane = tid & 31;
  const int h    = lane >> 4;   // K-half selector (A/B), row-half selector (C/D)
  const int cl   = lane & 15;   // A: row within tile; C/D: column within tile
  _Float16* myh1 = &h1s[wave * 16 * EH];
  const uint32_t* h1u = (const uint32_t*)myh1;

  for (int t = 0; t < 4; ++t) {
    const int jt = wave * 4 + t;           // tile 0..15
    const int s0 = 2 * jt;                 // 2 samples per tile
    // A-fragment row owned by this lane: m = cl -> (sample, agent i)
    const int s_row  = s0 + (cl >> 3);
    const int i_row  = cl & 7;
    const int rank_i = (int)a2p_s[s_row * NAG + i_row];
    const uint8_t* perm_row = &p2a_s[s_row * NAG];

    // ---- build layer-1 A fragments (coalition features, K padded to 96) ----
    v16h afr[3];
#pragma unroll
    for (int ks = 0; ks < 3; ++ks) {
      v16h a;
#pragma unroll
      for (int v = 0; v < 8; ++v) {
        int kp = ks * 32 + h * 16 + 2 * v;
        float x0 = 0.f, x1 = 0.f;
        if (kp < K1) {
          int n0 = kp / AACT, a0 = kp % AACT;
          if (n0 <= rank_i) x0 = aq[a0 * NAG + (int)perm_row[n0]];
        }
        if (kp + 1 < K1) {
          int n1 = (kp + 1) / AACT, a1 = (kp + 1) % AACT;
          if (n1 <= rank_i) x1 = aq[a1 * NAG + (int)perm_row[n1]];
        }
        a[2 * v]     = (_Float16)x0;
        a[2 * v + 1] = (_Float16)x1;
      }
      afr[ks] = a;
    }

    // ---- layer 1: h1 = relu(x @ W1q + sp + b1), C-init = spb1 broadcast ----
    v8f acc1[4];
#pragma unroll
    for (int nt = 0; nt < 4; ++nt) {
      float init = spb1[nt * 16 + cl];
#pragma unroll
      for (int q = 0; q < 8; ++q) acc1[nt][q] = init;
#pragma unroll
      for (int ks = 0; ks < 3; ++ks) {
        v16h bfr = __builtin_bit_cast(v16h, w1f[(ks * 4 + nt) * 32 + lane]);
        acc1[nt] = __builtin_amdgcn_wmma_f32_16x16x32_f16(
            false, afr[ks], false, bfr, (short)0, acc1[nt], false, false);
      }
    }

    // ---- relu + transpose bounce through LDS (C layout -> A layout) ----
#pragma unroll
    for (int nt = 0; nt < 4; ++nt)
#pragma unroll
      for (int q = 0; q < 8; ++q) {
        float x = acc1[nt][q];
        x = x > 0.f ? x : 0.f;
        myh1[(q + 8 * h) * EH + nt * 16 + cl] = (_Float16)x;
      }
    asm volatile("s_wait_dscnt 0" ::: "memory");

    // ---- layer 2: h2 = relu(h1 @ W2 + b2), C-init = b2 broadcast ----
    v8f acc2[4];
#pragma unroll
    for (int nt = 0; nt < 4; ++nt) {
      float init = b2s[nt * 16 + cl];
#pragma unroll
      for (int q = 0; q < 8; ++q) acc2[nt][q] = init;
    }
#pragma unroll
    for (int ks = 0; ks < 2; ++ks) {
      v16h a;
      uint32_t* au = (uint32_t*)&a;
#pragma unroll
      for (int v = 0; v < 8; ++v) {
        int kp = ks * 32 + h * 16 + 2 * v;
        au[v] = h1u[(cl * EH + kp) >> 1];   // kp even: one dword = 2 halves
      }
#pragma unroll
      for (int nt = 0; nt < 4; ++nt) {
        v16h bfr = __builtin_bit_cast(v16h, w2f[(ks * 4 + nt) * 32 + lane]);
        acc2[nt] = __builtin_amdgcn_wmma_f32_16x16x32_f16(
            false, a, false, bfr, (short)0, acc2[nt], false, false);
      }
    }

    // ---- advs = relu(h2) @ Wa : per-lane partials, reduce across 16 cols ----
    float part[8];
#pragma unroll
    for (int q = 0; q < 8; ++q) part[q] = 0.f;
#pragma unroll
    for (int nt = 0; nt < 4; ++nt) {
      float w = was[nt * 16 + cl];
#pragma unroll
      for (int q = 0; q < 8; ++q) {
        float x = acc2[nt][q];
        x = x > 0.f ? x : 0.f;
        part[q] += x * w;
      }
    }
#pragma unroll
    for (int q = 0; q < 8; ++q) {
      float p = part[q];
      p += __shfl_xor(p, 1, 32);
      p += __shfl_xor(p, 2, 32);
      p += __shfl_xor(p, 4, 32);
      p += __shfl_xor(p, 8, 32);
      part[q] = p;
    }
    // lane cl==0 of each half owns rows m = q + 8h (i = q, sample = s0 + h)
    if (cl == 0) {
#pragma unroll
      for (int q = 0; q < 8; ++q) atomicAdd(&out_acc[q], part[q]);
    }
  }

  __syncthreads();
  if (tid < NAG) {
    out[b * NAG + tid] = out_acc[tid] * (1.f / (float)SSAMP) + ba[0] + v_g[b];
  }
}

// ---------------------------------------------------------------------------
extern "C" void kernel_launch(void* const* d_in, const int* in_sizes, int n_in,
                              void* d_out, int out_size, void* d_ws, size_t ws_size,
                              hipStream_t stream) {
  const float* states  = (const float*)d_in[0];
  const float* agentqs = (const float*)d_in[1];
  const float* W1  = (const float*)d_in[2];
  const float* b1  = (const float*)d_in[3];
  const float* W2  = (const float*)d_in[4];
  const float* b2  = (const float*)d_in[5];
  const float* Wa  = (const float*)d_in[6];
  const float* ba  = (const float*)d_in[7];
  const float* Wv1 = (const float*)d_in[8];
  const float* bv1 = (const float*)d_in[9];
  const float* Wv2 = (const float*)d_in[10];
  const float* bv2 = (const float*)d_in[11];
  const int*   rng = (const int*)d_in[12];
  float* out = (float*)d_out;

  // workspace layout (all 256B-aligned)
  uint8_t* ws = (uint8_t*)d_ws;
  uint8_t* p2a   = ws;                              // 262144 B
  uint8_t* a2p   = ws + (256u << 10);               // 262144 B
  float*   spb1  = (float*)(ws + (512u << 10));     // 262144 B
  float*   vbuf  = (float*)(ws + (768u << 10));     //   4096 B
  uint32_t* w1p  = (uint32_t*)(ws + (772u << 10));  //  12288 B
  uint32_t* w2p  = (uint32_t*)(ws + (784u << 10));  //   8192 B

  perm_kernel<<<(BT * SSAMP + 255) / 256, 256, 0, stream>>>(rng, p2a, a2p);
  state_kernel<<<BT, EH, 0, stream>>>(states, W1, b1, Wv1, bv1, Wv2, bv2, spb1, vbuf);
  pack_kernel<<<3, 256, 0, stream>>>(W1, W2, w1p, w2p);
  mixer_main<<<BT, 128, 0, stream>>>(agentqs, b2, Wa, ba, p2a, a2p, spb1, vbuf,
                                     (const v8u*)w1p, (const v8u*)w2p, out);
}
